// CFConv_82420422410614
// MI455X (gfx1250) — compile-verified, matrix operand order checked by reading, not run on hardware
//
#include <hip/hip_runtime.h>
#include <hip/hip_fp16.h>
#include <math.h>

typedef __attribute__((ext_vector_type(16))) _Float16 v16h;
typedef __attribute__((ext_vector_type(8)))  float    v8f;

#define LOG2F 0.6931471805599453f

// CDNA5 split dependency counter: wait for all outstanding LDS ops from this
// wave before reading another lane's LDS data (intra-wave staging fence).
__device__ __forceinline__ void wave_lds_fence() {
  asm volatile("s_wait_dscnt 0" ::: "memory");
}

// ---- WMMA fragment layout helpers (ISA 7.12.2, wave32) ------------------
// A fragment: 16x32 f16 (M x K), row-major tile in LDS, row stride `ld` halves.
// Per lane the 16 halves are two contiguous 8-half runs -> ds_load_b128 x2.
__device__ __forceinline__ v16h load_frag_a(const _Float16* base, int ld) {
  int lane = threadIdx.x & 31;
  int m = lane & 15;
  int kb = (lane >> 4) << 3;
  v16h a;
#pragma unroll
  for (int v = 0; v < 8; ++v) {
    int ko = kb + ((v & 4) << 2) + ((v & 3) << 1);
    a[2 * v]     = base[m * ld + ko];
    a[2 * v + 1] = base[m * ld + ko + 1];
  }
  return a;
}

// B fragment (32x16, K x N): inverse map used to PRE-SWIZZLE weights into
// fragment-major LDS so the hot-loop load is a single aligned 32B read.
__device__ __forceinline__ int bfrag_offset(int kk, int n) {
  int hi   = (kk >> 3) & 1;
  int krem = kk - (hi << 3);       // 0..7 or 16..23
  int b2   = (krem >> 4) & 1;
  int v2   = (krem >> 1) & 3;
  int p    = kk & 1;
  int lane = (hi << 4) | n;
  int slot = (b2 << 3) | (v2 << 1) | p;
  return lane * 16 + slot;         // within one 512-half tile
}

// ---- zero-init (agg must be zeroed every launch; harness poisons ws) ----
__global__ void zero_f32(float4* __restrict__ p, int n4) {
  int i = blockIdx.x * blockDim.x + threadIdx.x;
  int stride = gridDim.x * blockDim.x;
  float4 z = {0.f, 0.f, 0.f, 0.f};
  for (; i < n4; i += stride) p[i] = z;
}

// ---- out[r,:] = in[r,:] @ w^T + bias  (128x128 weight), WMMA ------------
__global__ __launch_bounds__(256) void node_linear_wmma(
    const float* __restrict__ in, const float* __restrict__ w,
    const float* __restrict__ bias, float* __restrict__ out, int nrows) {
  __shared__ v16h     BwF[8 * 4 * 32];   // fragment-major weights (32KB)
  __shared__ _Float16 Ax[8][16 * 128];   // per-wave A staging (32KB)
  int tid = threadIdx.x;
  _Float16* BwH = (_Float16*)BwF;
  for (int i = tid; i < 128 * 128; i += 256) {
    int f = i >> 7, k = i & 127;                   // w[i] coalesced
    int ks = k >> 5, kk = k & 31, ft = f >> 4, n = f & 15;
    BwH[(((ft << 2) | ks) << 9) + bfrag_offset(kk, n)] = (_Float16)w[i];
  }
  __syncthreads();

  int wave = tid >> 5, lane = tid & 31;
  int m0 = (blockIdx.x * 8 + wave) * 16;
  if (m0 >= nrows) return;
  int hi = lane >> 4, n = lane & 15;
  bool full = (m0 + 16 <= nrows);

  if (full) {
    for (int i = lane; i < 16 * 128; i += 32)
      Ax[wave][i] = (_Float16)in[m0 * 128 + i];
  } else {
    int mrows = nrows - m0;
    for (int i = lane; i < 16 * 128; i += 32) {
      int r = i >> 7;
      Ax[wave][i] = (r < mrows) ? (_Float16)in[m0 * 128 + i] : (_Float16)0.f;
    }
  }
  wave_lds_fence();

#pragma unroll
  for (int ft = 0; ft < 8; ++ft) {
    v8f acc = {};
#pragma unroll
    for (int ks = 0; ks < 4; ++ks) {
      v16h a = load_frag_a(&Ax[wave][ks * 32], 128);
      v16h b = BwF[(((ft << 2) | ks) << 5) + lane];  // one aligned 32B LDS read
      acc = __builtin_amdgcn_wmma_f32_16x16x32_f16(false, a, false, b,
                                                   (short)0, acc, false, false);
    }
    int c = ft * 16 + n;
    float bv = bias[c];
    if (full) {
      float* p = out + (m0 + hi * 8) * 128 + c;  // shared base, imm offsets
#pragma unroll
      for (int v = 0; v < 8; ++v) p[v * 128] = acc[v] + bv;
    } else {
#pragma unroll
      for (int v = 0; v < 8; ++v) {
        int m = v + hi * 8;
        if (m0 + m < nrows) out[(m0 + m) * 128 + c] = acc[v] + bv;
      }
    }
  }
}

// ---- fused edge pipeline: filter-net (2 WMMA GEMMs + ssp) * gather + scatter
// Persistent workgroups: weights staged once, grid-stride over 16-edge tiles.
__global__ __launch_bounds__(256) void cfconv_edge_wmma(
    const float* __restrict__ edge_attr, const long long* __restrict__ edge_index,
    const float* __restrict__ fn1_w, const float* __restrict__ fn1_b,
    const float* __restrict__ fn2_w, const float* __restrict__ fn2_b,
    const float* __restrict__ h, float* __restrict__ agg, int E) {
  __shared__ v16h     F1F[8 * 2 * 32];   // fn1_w fragment-major (16KB)
  __shared__ v16h     F2F[8 * 4 * 32];   // fn2_w fragment-major (32KB)
  __shared__ _Float16 EA[8][16 * 64];    // per-wave edge_attr tile (16KB)
  __shared__ _Float16 T1[8][16 * 128];   // per-wave ssp(GEMM1) restaged (32KB)
  int tid = threadIdx.x;
  _Float16* F1H = (_Float16*)F1F;
  _Float16* F2H = (_Float16*)F2F;
  for (int i = tid; i < 64 * 128; i += 256) {
    int f = i >> 6, k = i & 63;
    int ks = k >> 5, kk = k & 31, ft = f >> 4, n = f & 15;
    float v = (k < 50) ? fn1_w[f * 50 + k] : 0.f;
    F1H[(((ft << 1) | ks) << 9) + bfrag_offset(kk, n)] = (_Float16)v;
  }
  for (int i = tid; i < 128 * 128; i += 256) {
    int f = i >> 7, k = i & 127;
    int ks = k >> 5, kk = k & 31, ft = f >> 4, n = f & 15;
    F2H[(((ft << 2) | ks) << 9) + bfrag_offset(kk, n)] = (_Float16)fn2_w[i];
  }
  __syncthreads();

  int wave = tid >> 5, lane = tid & 31;
  int hi = lane >> 4, n = lane & 15;

  // per-lane bias values, hoisted out of the tile loop
  float fb1[8], fb2[8];
#pragma unroll
  for (int ft = 0; ft < 8; ++ft) {
    fb1[ft] = fn1_b[ft * 16 + n];
    fb2[ft] = fn2_b[ft * 16 + n];
  }

  int tiles = (E + 15) >> 4;
  int tstep = gridDim.x * 8;

  for (int t = blockIdx.x * 8 + wave; t < tiles; t += tstep) {
    int e0 = t * 16;
    bool full = (e0 + 16 <= E);

    // prefetch next tile's edge_attr (3200B), one cacheline per low lane
    int tn = t + tstep;
    if (tn < tiles && lane < 13)
      __builtin_prefetch((const void*)(edge_attr + tn * 16 * 50 + lane * 64), 0, 1);

    // stage edge_attr [16 x 50] -> f16 padded [16 x 64]
    if (full) {
      for (int i = lane; i < 16 * 64; i += 32) {
        int r = i >> 6, c = i & 63;
        float v = (c < 50) ? edge_attr[e0 * 50 + r * 50 + c] : 0.f;
        EA[wave][i] = (_Float16)v;
      }
    } else {
      for (int i = lane; i < 16 * 64; i += 32) {
        int r = i >> 6, c = i & 63;
        float v = (c < 50 && e0 + r < E) ? edge_attr[e0 * 50 + r * 50 + c] : 0.f;
        EA[wave][i] = (_Float16)v;
      }
    }
    wave_lds_fence();

    // GEMM1 + shifted softplus -> T1 (f16)
#pragma unroll
    for (int ft = 0; ft < 8; ++ft) {
      v8f acc = {};
#pragma unroll
      for (int ks = 0; ks < 2; ++ks) {
        v16h a = load_frag_a(&EA[wave][ks * 32], 64);
        v16h b = F1F[(((ft << 1) | ks) << 5) + lane];
        acc = __builtin_amdgcn_wmma_f32_16x16x32_f16(false, a, false, b,
                                                     (short)0, acc, false, false);
      }
      float bv = fb1[ft];
#pragma unroll
      for (int v = 0; v < 8; ++v) {
        float tv = acc[v] + bv;
        float s = (tv > 20.f) ? tv : __logf(1.f + __expf(tv));  // softplus
        int m = v + hi * 8;
        T1[wave][m * 128 + ft * 16 + n] = (_Float16)(s - LOG2F);
      }
    }
    wave_lds_fence();

    // per-lane gather/scatter row offsets (32-bit, GVS scale_offset friendly)
    int roff[8], coff[8];
#pragma unroll
    for (int v = 0; v < 8; ++v) {
      int e = e0 + v + hi * 8;
      int ec = (e < E) ? e : e0;
      roff[v] = ((int)edge_index[ec]) << 7;      // destination row * 128
      coff[v] = ((int)edge_index[E + ec]) << 7;  // source row * 128
    }

    // GEMM2 per 16-channel tile, fused bias + gather-multiply + atomic scatter
#pragma unroll
    for (int ft = 0; ft < 8; ++ft) {
      v8f acc = {};
#pragma unroll
      for (int ks = 0; ks < 4; ++ks) {
        v16h a = load_frag_a(&T1[wave][ks * 32], 128);
        v16h b = F2F[(((ft << 2) | ks) << 5) + lane];
        acc = __builtin_amdgcn_wmma_f32_16x16x32_f16(false, a, false, b,
                                                     (short)0, acc, false, false);
      }
      float bv = fb2[ft];
      int c = ft * 16 + n;
      if (full) {
#pragma unroll
        for (int v = 0; v < 8; ++v) {
          float Wv = acc[v] + bv;
          float xv = h[coff[v] + c];                       // L2-resident gather
          unsafeAtomicAdd(&agg[roff[v] + c], Wv * xv);     // L2 fp32 atomic
        }
      } else {
#pragma unroll
        for (int v = 0; v < 8; ++v) {
          if (e0 + v + hi * 8 < E) {
            float Wv = acc[v] + bv;
            float xv = h[coff[v] + c];
            unsafeAtomicAdd(&agg[roff[v] + c], Wv * xv);
          }
        }
      }
    }
  }
}

extern "C" void kernel_launch(void* const* d_in, const int* in_sizes, int n_in,
                              void* d_out, int out_size, void* d_ws, size_t ws_size,
                              hipStream_t stream) {
  const float*     x          = (const float*)d_in[0];
  const long long* edge_index = (const long long*)d_in[1];  // int64 [2,E]
  const float*     edge_attr  = (const float*)d_in[2];
  const float*     lin1_w     = (const float*)d_in[3];
  const float*     lin1_b     = (const float*)d_in[4];
  const float*     lin2_w     = (const float*)d_in[5];
  const float*     lin2_b     = (const float*)d_in[6];
  const float*     fn1_w      = (const float*)d_in[7];
  const float*     fn1_b      = (const float*)d_in[8];
  const float*     fn2_w      = (const float*)d_in[9];
  const float*     fn2_b      = (const float*)d_in[10];

  int N = in_sizes[0] / 128;  // 100000
  int E = in_sizes[2] / 50;   // 1600000

  // workspace: h [N,128] f32, agg [N,128] f32  (102.4 MB total)
  float* h   = (float*)d_ws;
  float* agg = h + (size_t)N * 128;
  float* out = (float*)d_out;

  zero_f32<<<1024, 256, 0, stream>>>((float4*)agg, N * 32);

  int ngrid = ((N + 15) / 16 + 7) / 8;
  node_linear_wmma<<<ngrid, 256, 0, stream>>>(x, lin1_w, lin1_b, h, N);

  int etiles = (E + 15) / 16;
  int ewgs = (etiles + 7) / 8;
  int egrid = ewgs < 2048 ? ewgs : 2048;
  cfconv_edge_wmma<<<egrid, 256, 0, stream>>>(edge_attr, edge_index, fn1_w, fn1_b,
                                              fn2_w, fn2_b, h, agg, E);

  node_linear_wmma<<<ngrid, 256, 0, stream>>>(agg, lin2_w, lin2_b, out, N);
}
